// Deep_OSTL_Model_25829933318902
// MI455X (gfx1250) — matrix operand order repaired and checked by reading, not compiled, not run on hardware
//
#include <hip/hip_runtime.h>

// ---------------------------------------------------------------------------
// Deep OSTL model, fully fused for MI455X (gfx1250, wave32, WMMA).
//   z = x @ projW^T + proj_b
//   for l in 0..3:  zx = z @ Wx_l^T + bz_l ; h=0 ; 8x: h = tanh(h @ Wz_l^T + zx)
//   out = z @ headW^T + head_b
// 34 GEMMs of [32768,512]x[512,512] ~ 585 GFLOP, AI ~4400 FLOP/B -> compute
// bound: bf16 WMMA w/ f32 accum, activations resident in LDS per batch tile.
// Weight K-slabs double-buffered in LDS via async global->LDS copies.
// ---------------------------------------------------------------------------

typedef __bf16 v16bf __attribute__((ext_vector_type(16)));
typedef __bf16 v8bf  __attribute__((ext_vector_type(8)));
typedef float  v8f   __attribute__((ext_vector_type(8)));
typedef int    v4i   __attribute__((ext_vector_type(4)));

typedef __attribute__((address_space(1))) v4i* gv4i_p;   // global int4*
typedef __attribute__((address_space(3))) v4i* lv4i_p;   // LDS    int4*

#define H        512
#define BM       64            // batch rows per workgroup
#define BK       32            // K-chunk (matches 16x16x32 bf16 WMMA)
#define NKC      (H / BK)      // 16 K-chunks
#define THREADS  256           // 8 waves (wave32)
#define NMAT     (H * H)       // 262144 elements per weight matrix
#define SLAB     (BK * H)      // 16384 bf16 = 32KB per weight K-slab
#define BATCH    32768
#define NLAYERS  4
#define NSTEPS   8

#if defined(__has_builtin)
# if __has_builtin(__builtin_amdgcn_global_load_async_to_lds_b128)
#  define HAVE_ASYNC_LDS 1
# endif
#endif
#ifndef HAVE_ASYNC_LDS
# define HAVE_ASYNC_LDS 0
#endif

__device__ __forceinline__ float fast_tanh(float x) {
#if __has_builtin(__builtin_amdgcn_tanhf)
    return __builtin_amdgcn_tanhf(x);     // gfx1250 TRANS v_tanh_f32
#else
    return tanhf(x);
#endif
}

// A fragment per ISA 16-bit A layout (16x32):
//   lanes 0-15 : V0-3 = K 0..7,  V4-7 = K 16..23   (row M = lane)
//   lanes16-31 : V0-3 = K 8..15, V4-7 = K 24..31   (row M = lane-16)
__device__ __forceinline__ v16bf make_a_frag(const __bf16* p) {
    v8bf lo = *(const v8bf*)(p);
    v8bf hi = *(const v8bf*)(p + 16);
    v16bf f;
#pragma unroll
    for (int i = 0; i < 8; ++i) { f[i] = lo[i]; f[i + 8] = hi[i]; }
    return f;
}

// Stage one 32KB weight K-slab (contiguous in pre-transposed layout) into LDS.
// Async path: CDNA5 global_load_async_to_lds_b128 (ASYNCcnt), 16B/lane/op.
__device__ __forceinline__ void stage_slab(const __bf16* __restrict__ gsrc,
                                           __bf16* ldst, int tid)
{
#if HAVE_ASYNC_LDS
#pragma unroll
    for (int j = 0; j < (SLAB * 2) / (16 * THREADS); ++j) {   // 8 ops/thread
        const int idx = tid + j * THREADS;
        gv4i_p g = (gv4i_p)((__attribute__((address_space(1))) char*)gsrc
                            + (size_t)idx * 16);
        lv4i_p l = (lv4i_p)((__attribute__((address_space(3))) char*)ldst
                            + idx * 16);
        __builtin_amdgcn_global_load_async_to_lds_b128(g, l, 0, 0);
    }
#else
    const uint4* s = (const uint4*)gsrc;
    uint4* d = (uint4*)ldst;
#pragma unroll
    for (int j = 0; j < (SLAB * 2) / (16 * THREADS); ++j) {
        const int idx = tid + j * THREADS;
        d[idx] = s[idx];
    }
    __builtin_prefetch(gsrc + SLAB + (size_t)tid * 64, 0, 0);  // global_prefetch_b8
#endif
}

__device__ __forceinline__ void wait_stage() {
#if HAVE_ASYNC_LDS
# if __has_builtin(__builtin_amdgcn_s_wait_asynccnt)
    __builtin_amdgcn_s_wait_asynccnt(0);
# else
    asm volatile("s_wait_asynccnt 0x0" ::: "memory");
# endif
#endif
}

// 64x512 = 64x512 * (512x512)^T, K=512. WT is pre-transposed bf16 [K][N]
// (i.e. the B matrix row-major). Wave computes 2 M-tiles x 8 N-tiles so each
// B fragment is shared by two WMMAs (halves LDS reads per matrix op).
// B fragment per ISA B layout: N = lane&15, elements = K (same order as A).
__device__ __forceinline__ void gemm_tile(const __bf16* __restrict__ WT,
                                          const __bf16* act, __bf16* wtile,
                                          v8f* acc, int row0, int row1, int kbase,
                                          int wn, int lane, int tid)
{
#pragma unroll
    for (int t = 0; t < 16; ++t) {
        v8f z = {0.f, 0.f, 0.f, 0.f, 0.f, 0.f, 0.f, 0.f};
        acc[t] = z;
    }

    stage_slab(WT, wtile, tid);                   // slab 0 -> buf 0
    wait_stage();
    __syncthreads();

    for (int kc = 0; kc < NKC; ++kc) {
        const __bf16* cur = wtile + (kc & 1) * SLAB;
        if (kc + 1 < NKC)                         // overlap next slab w/ compute
            stage_slab(WT + (size_t)(kc + 1) * SLAB,
                       wtile + ((kc + 1) & 1) * SLAB, tid);

        v16bf a0 = make_a_frag(act + (size_t)row0 * H + kc * BK + kbase);
        v16bf a1 = make_a_frag(act + (size_t)row1 * H + kc * BK + kbase);
        const __bf16* pb0 = cur + (size_t)kbase * H + wn * 128 + (lane & 15);
#pragma unroll
        for (int nt = 0; nt < 8; ++nt) {
            v16bf bf;  // strided LDS gather over K
#pragma unroll
            for (int i = 0; i < 8; ++i) {
                bf[i]     = pb0[nt * 16 + (size_t)i * H];
                bf[i + 8] = pb0[nt * 16 + (size_t)(16 + i) * H];
            }
            acc[nt]     = __builtin_amdgcn_wmma_f32_16x16x32_bf16(
                false, a0, false, bf, (short)0, acc[nt], false, false);
            acc[8 + nt] = __builtin_amdgcn_wmma_f32_16x16x32_bf16(
                false, a1, false, bf, (short)0, acc[8 + nt], false, false);
        }
        wait_stage();
        __syncthreads();   // next buf ready; cur free for overwrite next iter
    }
    // final __syncthreads above also means all reads of `act` are complete,
    // so the caller may overwrite act in place.
}

__global__ __launch_bounds__(THREADS)
void ostl_fused(const float* __restrict__ x,
                const float* __restrict__ proj_b,
                const float* __restrict__ bz,
                const float* __restrict__ head_b,
                const __bf16* __restrict__ wsW,   // [10][512][512] bf16, K-major
                float* __restrict__ out)
{
    extern __shared__ __bf16 smem[];
    __bf16* act   = smem;                 // [BM][H] activations (bf16)
    __bf16* zx    = smem + BM * H;        // [BM][H] layer-invariant zx (bf16)
    __bf16* wtile = smem + 2 * BM * H;    // [2][BK][H] weight slabs (bf16)

    const int tid  = threadIdx.x;
    const int wave = tid >> 5;
    const int lane = tid & 31;
    const int wmg  = wave & 1;            // M group: tiles {2*wmg, 2*wmg+1}
    const int wn   = wave >> 1;           // N quarter 0..3 (8 N-tiles each)
    const int m0   = blockIdx.x * BM;

    const int kbase = (lane >> 4) * 8;
    const int row0  = (wmg * 2 + 0) * 16 + (lane & 15);
    const int row1  = (wmg * 2 + 1) * 16 + (lane & 15);
    // C/D layout: lanes 0-15 -> M=v, lanes 16-31 -> M=v+8 ; N = lane&15
    const int ecol  = wn * 128 + (lane & 15);

    const __bf16* Wproj = wsW;
    const __bf16* Wz    = wsW + (size_t)1 * NMAT;
    const __bf16* Wx    = wsW + (size_t)5 * NMAT;
    const __bf16* Whead = wsW + (size_t)9 * NMAT;

    // ---- load x tile, convert to bf16 in LDS ----
    for (int i = tid * 4; i < BM * H; i += THREADS * 4) {
        float4 v = *(const float4*)(x + (size_t)m0 * H + i);
        act[i + 0] = (__bf16)v.x; act[i + 1] = (__bf16)v.y;
        act[i + 2] = (__bf16)v.z; act[i + 3] = (__bf16)v.w;
    }
    __syncthreads();

    v8f acc[16];   // [a][nt] : 2 M-tiles x 8 N-tiles

    // ---- z = x @ projW^T + proj_b ----
    gemm_tile(Wproj, act, wtile, acc, row0, row1, kbase, wn, lane, tid);
#pragma unroll
    for (int a = 0; a < 2; ++a) {
        const int rb = (wmg * 2 + a) * 16 + (lane >> 4) * 8;
#pragma unroll
        for (int nt = 0; nt < 8; ++nt) {
            const int c = ecol + nt * 16;
            const float b = proj_b[c];
#pragma unroll
            for (int v = 0; v < 8; ++v)
                act[(rb + v) * H + c] = (__bf16)(acc[a * 8 + nt][v] + b);
        }
    }
    __syncthreads();

    // ---- layers ----
    for (int l = 0; l < NLAYERS; ++l) {
        // zx = z @ Wx^T + bz  (invariant across the 8 recurrence steps)
        gemm_tile(Wx + (size_t)l * NMAT, act, wtile, acc, row0, row1, kbase, wn, lane, tid);
#pragma unroll
        for (int a = 0; a < 2; ++a) {
            const int rb = (wmg * 2 + a) * 16 + (lane >> 4) * 8;
#pragma unroll
            for (int nt = 0; nt < 8; ++nt) {
                const int c = ecol + nt * 16;
                const float b = bz[l * H + c];
#pragma unroll
                for (int v = 0; v < 8; ++v)
                    zx[(rb + v) * H + c] = (__bf16)(acc[a * 8 + nt][v] + b);
            }
        }
        __syncthreads();

        // step 1: h0 = 0  =>  h1 = tanh(zx)  (skip the zero GEMM)
        for (int i = tid; i < BM * H; i += THREADS)
            act[i] = (__bf16)fast_tanh((float)zx[i]);
        __syncthreads();

        // steps 2..8: h = tanh(h @ Wz^T + zx)
        for (int s = 1; s < NSTEPS; ++s) {
            gemm_tile(Wz + (size_t)l * NMAT, act, wtile, acc, row0, row1, kbase, wn, lane, tid);
#pragma unroll
            for (int a = 0; a < 2; ++a) {
                const int rb = (wmg * 2 + a) * 16 + (lane >> 4) * 8;
#pragma unroll
                for (int nt = 0; nt < 8; ++nt) {
                    const int c = ecol + nt * 16;
#pragma unroll
                    for (int v = 0; v < 8; ++v) {
                        const int r = rb + v;
                        act[r * H + c] =
                            (__bf16)fast_tanh(acc[a * 8 + nt][v] + (float)zx[r * H + c]);
                    }
                }
            }
            __syncthreads();
        }
    }

    // ---- out = h @ headW^T + head_b  (f32 to global) ----
    gemm_tile(Whead, act, wtile, acc, row0, row1, kbase, wn, lane, tid);
#pragma unroll
    for (int a = 0; a < 2; ++a) {
        const int rb = (wmg * 2 + a) * 16 + (lane >> 4) * 8;
#pragma unroll
        for (int nt = 0; nt < 8; ++nt) {
            const int c = ecol + nt * 16;
            const float b = head_b[c];
#pragma unroll
            for (int v = 0; v < 8; ++v)
                out[(size_t)(m0 + rb + v) * H + c] = acc[a * 8 + nt][v] + b;
        }
    }
}

// ---------------------------------------------------------------------------
// Prep: convert all f32 weights to bf16 AND transpose to K-major so the main
// kernel's B fragments / LDS staging are fully contiguous.
// ws layout: [0]=projW, [1..4]=Wz, [5..8]=Wx, [9]=headW ; each [K][N].
// ---------------------------------------------------------------------------
__global__ void convert_weights(const float* __restrict__ projW,
                                const float* __restrict__ Wz,
                                const float* __restrict__ Wx,
                                const float* __restrict__ headW,
                                __bf16* __restrict__ ws)
{
    const size_t S = NMAT;
    size_t i = (size_t)blockIdx.x * blockDim.x + threadIdx.x;
    if (i >= 10 * S) return;
    size_t m = i / S;                 // which matrix
    size_t j = i - m * S;             // dest element, row-major [k][n]
    size_t k = j >> 9, n = j & 511;
    size_t srcj = n * H + k;          // source is [n][k] row-major
    float v;
    if (m == 0)      v = projW[srcj];
    else if (m < 5)  v = Wz[(m - 1) * S + srcj];
    else if (m < 9)  v = Wx[(m - 5) * S + srcj];
    else             v = headW[srcj];
    ws[i] = (__bf16)v;
}

extern "C" void kernel_launch(void* const* d_in, const int* in_sizes, int n_in,
                              void* d_out, int out_size, void* d_ws, size_t ws_size,
                              hipStream_t stream)
{
    const float* x     = (const float*)d_in[0];
    const float* projW = (const float*)d_in[1];
    const float* projb = (const float*)d_in[2];
    const float* Wz    = (const float*)d_in[3];
    const float* bz    = (const float*)d_in[4];
    const float* Wx    = (const float*)d_in[5];
    const float* headW = (const float*)d_in[6];
    const float* headb = (const float*)d_in[7];
    __bf16* ws = (__bf16*)d_ws;       // needs 10*512*512*2 = 5.25 MB

    const int total = 10 * NMAT;
    convert_weights<<<(total + THREADS - 1) / THREADS, THREADS, 0, stream>>>(
        projW, Wz, Wx, headW, ws);

    const size_t shmem = (size_t)(2 * BM * H + 2 * SLAB) * sizeof(__bf16);  // 192 KB
    (void)hipFuncSetAttribute(reinterpret_cast<const void*>(ostl_fused),
                              hipFuncAttributeMaxDynamicSharedMemorySize, (int)shmem);
    ostl_fused<<<BATCH / BM, THREADS, shmem, stream>>>(
        x, projb, bz, headb, ws, (float*)d_out);
}
